// CPDFP_25701084299789
// MI455X (gfx1250) — matrix operand order repaired and checked by reading, not compiled, not run on hardware
//
#include <hip/hip_runtime.h>
#include <hip/hip_bf16.h>
#include <stdint.h>

// ---------------------------------------------------------------------------
// Problem: B=64, C=256, H=W=32 (HW=1024). out[b,c] = sum over branches of
//   (sum_hw x*attn) / (sum_hw attn + 1e-8),  attn = sigmoid(conv_w @ [x; dot])
// caLayer is identity (softmax over size-1 axis == 1).
// ---------------------------------------------------------------------------

#define CCH   256          // channels
#define HW    1024         // spatial
#define NKT   9            // K tiles of 32 (K padded 257 -> 288)
#define KPH   304          // bf16 LDS row stride (halves); mult of 16 (32B align)
#define KPF   260          // f32 LDS row stride (floats)
#define NSPLIT 4           // hw splits per (b,branch)

typedef __bf16 v16bf __attribute__((ext_vector_type(16)));
typedef float  v8f   __attribute__((ext_vector_type(8)));

union Frag16 { v16bf v; uint32_t u[8]; uint4 q[2]; };

// native f32 -> bf16 (RNE) conversion; lowers to hardware cvt on gfx1250
__device__ __forceinline__ unsigned short bfh(float f) {
    union { __bf16 h; unsigned short s; } u;
    u.h = (__bf16)f;
    return u.s;
}

// fast sigmoid: v_exp_f32 + v_rcp_f32 (~1ulp), well below bf16 GEMM noise
__device__ __forceinline__ float fast_sigmoid(float x) {
    return __builtin_amdgcn_rcpf(1.0f + __expf(-x));
}

// ---------------------------------------------------------------------------
// Kernel 1: pack conv_w (256 x 257 f32) into bf16 WMMA A-fragments in ws.
// Fragment (mt,kt): 32 lanes x 16 bf16 (32B/lane). k order (contiguous, same as B):
//   lane<16 : M=lane,   VGPR v holds K = {2v,2v+1} (v<8)
//   lane>=16: M=lane-16, same +16 on K.
// K index 256 is the "dot" channel weight; K in [257,288) is zero padding.
// ---------------------------------------------------------------------------
__global__ __launch_bounds__(32) void k_packA(const float* __restrict__ conv_w,
                                              uint32_t* __restrict__ wsA) {
    int frag = blockIdx.x;            // 0 .. 16*9-1
    int mt = frag / NKT, kt = frag % NKT;
    int lane = threadIdx.x;
    int M = mt * 16 + (lane & 15);
    int khi = (lane & 16) ? 16 : 0;
    uint32_t out[8];
#pragma unroll
    for (int v = 0; v < 8; ++v) {
        unsigned short h[2];
#pragma unroll
        for (int p = 0; p < 2; ++p) {
            int base = v * 2 + p;
            int k = kt * 32 + base + khi;
            float f = (k < CCH + 1) ? conv_w[M * (CCH + 1) + k] : 0.0f;
            h[p] = bfh(f);
        }
        out[v] = (uint32_t)h[0] | ((uint32_t)h[1] << 16);
    }
    uint4* dst = (uint4*)wsA + ((size_t)frag * 32 + lane) * 2;
    dst[0] = make_uint4(out[0], out[1], out[2], out[3]);
    dst[1] = make_uint4(out[4], out[5], out[6], out[7]);
}

// ---------------------------------------------------------------------------
// Kernel 2: dot[b,hw] = (1/C) * sum_c x[b,c,hw] * x[b,c,center]  per branch.
// ---------------------------------------------------------------------------
__global__ __launch_bounds__(256) void k_dot(const float* __restrict__ x1,
                                             const float* __restrict__ x2,
                                             float* __restrict__ dotbuf) {
    __shared__ float cen[CCH];
    int bb = blockIdx.x;              // branch*64 + b
    int branch = bb >> 6, b = bb & 63;
    const float* xb = (branch ? x2 : x1) + (size_t)b * CCH * HW;
    int t = threadIdx.x;
    cen[t] = xb[(size_t)t * HW + 528];            // center = (16,16) -> 16*32+16
    __syncthreads();
    float4 acc = make_float4(0.f, 0.f, 0.f, 0.f);
    for (int c = 0; c < CCH; ++c) {
        float4 v = *(const float4*)(xb + (size_t)c * HW + 4 * t);
        float w = cen[c];
        acc.x += v.x * w; acc.y += v.y * w; acc.z += v.z * w; acc.w += v.w * w;
    }
    const float inv = 1.0f / (float)CCH;
    *(float4*)(dotbuf + (size_t)bb * HW + 4 * t) =
        make_float4(acc.x * inv, acc.y * inv, acc.z * inv, acc.w * inv);
}

// ---------------------------------------------------------------------------
// Kernel 3: fused  attn = sigmoid(W @ [x;dot])  ->  S1=Σattn, S2=Σ x*attn.
// Block = (b, branch, split of 256 hw cols). 8 waves; wave owns M-tiles
// {2w, 2w+1}. LDS: bf16 slab [n][k] (k-contiguous; B frag = 2x ds_load_b128,
// zero VALU) + f32 slab [n][c] for the exact x values used by S2.
// B-fragment layout (32x16 bf16): lane<16: N=lane, K=kt*32+{0..15};
// lane>=16: N=lane-16, K=kt*32+16+{0..15}.
// B fragments are software-pipelined one K-step ahead so the ds_load latency
// is hidden under the previous step's two WMMAs.
// ---------------------------------------------------------------------------
__global__ __launch_bounds__(256) void k_gemm_pool(const float* __restrict__ x1,
                                                   const float* __restrict__ x2,
                                                   const uint32_t* __restrict__ wsA,
                                                   const float* __restrict__ dotbuf,
                                                   float* __restrict__ s1part,
                                                   float* __restrict__ s2part) {
    __shared__ __align__(16) unsigned short slabh[16 * KPH];  // bf16 [n][k]
    __shared__ float slabf[16 * KPF];                         // f32  [n][c]
    int blk = blockIdx.x;             // 0..511
    int split = blk & (NSPLIT - 1);
    int bb = blk >> 2;                // branch*64 + b
    int branch = bb >> 6, b = bb & 63;
    const float* xb = (branch ? x2 : x1) + (size_t)b * CCH * HW;

    int tid = threadIdx.x;
    int lane = tid & 31, wave = tid >> 5;
    int nlo = lane & 15, nhi = lane >> 4;
    int mt0 = wave * 2, mt1 = wave * 2 + 1;

    // zero bf16 pad region k in [257, KPH) once (never rewritten by fills)
    for (int i = tid; i < 16 * (KPH - (CCH + 1)); i += 256) {
        int n = i / (KPH - (CCH + 1));
        int k = (CCH + 1) + i % (KPH - (CCH + 1));
        slabh[n * KPH + k] = 0;
    }

    // hoist this wave's 18 A-fragments into registers (invariant over N loop)
    Frag16 aF0[NKT], aF1[NKT];
#pragma unroll
    for (int kt = 0; kt < NKT; ++kt) {
        const uint4* p0 = (const uint4*)wsA + ((size_t)(mt0 * NKT + kt) * 32 + lane) * 2;
        const uint4* p1 = (const uint4*)wsA + ((size_t)(mt1 * NKT + kt) * 32 + lane) * 2;
        aF0[kt].q[0] = p0[0]; aF0[kt].q[1] = p0[1];
        aF1[kt].q[0] = p1[0]; aF1[kt].q[1] = p1[1];
    }

    float s1a[8] = {0}, s2a[8] = {0}, s1b[8] = {0}, s2b[8] = {0};

    for (int t = 0; t < 16; ++t) {
        int n0 = split * 256 + t * 16;
        __syncthreads();                                   // prev slab consumed
        {   // fill: thread tid == channel c; convert to bf16 ONCE per element
            int c = tid;
            const float* src = xb + (size_t)c * HW + n0;
            float4 v0 = *(const float4*)(src + 0);
            float4 v1 = *(const float4*)(src + 4);
            float4 v2 = *(const float4*)(src + 8);
            float4 v3 = *(const float4*)(src + 12);
            float vv[16] = {v0.x, v0.y, v0.z, v0.w, v1.x, v1.y, v1.z, v1.w,
                            v2.x, v2.y, v2.z, v2.w, v3.x, v3.y, v3.z, v3.w};
#pragma unroll
            for (int j = 0; j < 16; ++j) {
                slabf[j * KPF + c] = vv[j];
                slabh[j * KPH + c] = bfh(vv[j]);
            }
            if (tid < 16)
                slabh[tid * KPH + CCH] = bfh(dotbuf[(size_t)bb * HW + n0 + tid]);
        }
        __syncthreads();

        // base of this lane's B fragments; K-step stride = 32 halves = 4 uint4
        const uint4* bbase = (const uint4*)&slabh[nlo * KPH + nhi * 16];

        v8f acc0 = {}, acc1 = {};
        Frag16 bcur, bnxt;
        bcur.q[0] = bbase[0]; bcur.q[1] = bbase[1];
#pragma unroll
        for (int kt = 0; kt < NKT; ++kt) {
            if (kt + 1 < NKT) {                       // prefetch next K-step
                bnxt.q[0] = bbase[(kt + 1) * 4];
                bnxt.q[1] = bbase[(kt + 1) * 4 + 1];
            }
            acc0 = __builtin_amdgcn_wmma_f32_16x16x32_bf16(
                       false, aF0[kt].v, false, bcur.v, (short)0, acc0, false, false);
            acc1 = __builtin_amdgcn_wmma_f32_16x16x32_bf16(
                       false, aF1[kt].v, false, bcur.v, (short)0, acc1, false, false);
            bcur = bnxt;
        }
        // sigmoid + fused pooling sums; C/D layout: VGPR g, lane l ->
        //   row m = g + 8*(l>>4), col n = l&15
#pragma unroll
        for (int g = 0; g < 8; ++g) {
            float a0 = fast_sigmoid(acc0[g]);
            float a1 = fast_sigmoid(acc1[g]);
            s1a[g] += a0;
            s2a[g] += a0 * slabf[nlo * KPF + (mt0 * 16 + g + 8 * nhi)];
            s1b[g] += a1;
            s2b[g] += a1 * slabf[nlo * KPF + (mt1 * 16 + g + 8 * nhi)];
        }
    }

    // reduce over the 16 columns (lanes sharing m): xor within half-wave
    int base = (bb * NSPLIT + split) * CCH;
#pragma unroll
    for (int g = 0; g < 8; ++g) {
        float r0 = s1a[g], r1 = s2a[g], r2 = s1b[g], r3 = s2b[g];
#pragma unroll
        for (int m = 1; m < 16; m <<= 1) {
            r0 += __shfl_xor(r0, m, 32);
            r1 += __shfl_xor(r1, m, 32);
            r2 += __shfl_xor(r2, m, 32);
            r3 += __shfl_xor(r3, m, 32);
        }
        if (nlo == 0) {                    // lanes 0 (m=g) and 16 (m=g+8)
            int m = nhi * 8 + g;
            s1part[base + mt0 * 16 + m] = r0;
            s2part[base + mt0 * 16 + m] = r1;
            s1part[base + mt1 * 16 + m] = r2;
            s2part[base + mt1 * 16 + m] = r3;
        }
    }
}

// ---------------------------------------------------------------------------
// Kernel 4: out[b,c] = Σ_branch ( Σ_split S2 ) / ( Σ_split S1 + 1e-8 )
// (exact IEEE division here — it is the final normalization)
// ---------------------------------------------------------------------------
__global__ __launch_bounds__(256) void k_finalize(const float* __restrict__ s1part,
                                                  const float* __restrict__ s2part,
                                                  float* __restrict__ out) {
    int b = blockIdx.x, c = threadIdx.x;
    float r = 0.0f;
#pragma unroll
    for (int branch = 0; branch < 2; ++branch) {
        int bb = branch * 64 + b;
        float s1 = 0.0f, s2 = 0.0f;
#pragma unroll
        for (int s = 0; s < NSPLIT; ++s) {
            int idx = (bb * NSPLIT + s) * CCH + c;
            s1 += s1part[idx];
            s2 += s2part[idx];
        }
        r += s2 / (s1 + 1e-8f);
    }
    out[b * CCH + c] = r;
}

// ---------------------------------------------------------------------------
// Workspace layout (bytes):
//   [0, 147456)              packed bf16 A fragments (16*9 frags * 1024B)
//   [147456, 671744)         dotbuf: 2*64*1024 f32
//   [671744, 1196032)        s1part: 2*64*4*256 f32
//   [1196032, 1720320)       s2part: 2*64*4*256 f32
// ---------------------------------------------------------------------------
extern "C" void kernel_launch(void* const* d_in, const int* in_sizes, int n_in,
                              void* d_out, int out_size, void* d_ws, size_t ws_size,
                              hipStream_t stream) {
    const float* x1     = (const float*)d_in[0];
    const float* x2     = (const float*)d_in[1];
    const float* conv_w = (const float*)d_in[2];
    // d_in[3..6] (ca layer) are mathematically unused: softmax over size-1 axis == 1.
    float* out = (float*)d_out;

    char* ws = (char*)d_ws;
    uint32_t* wsA   = (uint32_t*)(ws);
    float* dotbuf   = (float*)(ws + 147456);
    float* s1part   = (float*)(ws + 671744);
    float* s2part   = (float*)(ws + 1196032);

    k_packA<<<16 * NKT, 32, 0, stream>>>(conv_w, wsA);
    k_dot<<<128, 256, 0, stream>>>(x1, x2, dotbuf);
    k_gemm_pool<<<128 * NSPLIT, 256, 0, stream>>>(x1, x2, wsA, dotbuf, s1part, s2part);
    k_finalize<<<64, 256, 0, stream>>>(s1part, s2part, out);
}
